// ODERNNVAEEncoder_11922829214475
// MI455X (gfx1250) — matrix operand order repaired and checked by reading, not compiled, not run on hardware
//
#include <hip/hip_runtime.h>

// ---------------------------------------------------------------------------
// ODE-RNN VAE encoder for MI455X (gfx1250, wave32, WMMA).
// GEMMs: v_wmma_f32_16x16x32_bf16, double-buffered LDS staged with
// global_load_async_to_lds_b128 (ASYNCcnt) so global traffic overlaps WMMA.
// Staging addresses are pointer-stepped (constant stride per K-tile) to keep
// the hot loop free of 64-bit multiplies.
// Weights fp32->bf16 once per launch; activations bf16; z/h state f32.
// ---------------------------------------------------------------------------

typedef __attribute__((ext_vector_type(16))) __bf16 v16bf;
typedef __attribute__((ext_vector_type(8)))  float  v8f;

#define LEAKY 0.1f

// fp32 -> bf16 round-to-nearest-even
__device__ __forceinline__ unsigned short f2bf(float f) {
    unsigned int u = __float_as_uint(f);
    u += 0x7FFFu + ((u >> 16) & 1u);
    return (unsigned short)(u >> 16);
}

__device__ __forceinline__ float sigmoidf_(float x) {
    return 1.0f / (1.0f + expf(-x));
}

// ---- CDNA5 async global->LDS copy (no VGPR round-trip, tracked by ASYNCcnt)
__device__ __forceinline__ void async_ld_b128(unsigned lds_addr,
                                              const unsigned short* gptr) {
    asm volatile("global_load_async_to_lds_b128 %0, %1, off"
                 :: "v"(lds_addr), "v"(gptr)
                 : "memory");
}
__device__ __forceinline__ void wait_async_le4() {
    asm volatile("s_wait_asynccnt 4" ::: "memory");
}
__device__ __forceinline__ void wait_async_0() {
    asm volatile("s_wait_asynccnt 0" ::: "memory");
}
// flat shared pointer -> 32-bit LDS byte address (low 32 bits of the address)
__device__ __forceinline__ unsigned lds_addr_of(const void* p) {
    return (unsigned)(size_t)p;
}

// ---------------------------------------------------------------------------
// WMMA GEMM: C[M,N] = A[M,K] (bf16, row-major) x B[K,N] (bf16, row-major) + bias
// EPI == 0 : store f32 ; EPI == 1 : store bf16(tanh(acc + bias))
// Block tile 128x128, BK=32, 256 threads (8 waves). Wave grid 2x4; each wave
// computes a 64x32 region = 4x2 tiles of 16x16. Double-buffered async LDS.
// ---------------------------------------------------------------------------
#define BM 128
#define BN 128
#define BK 32
#define LDA_S (BK + 8)     // padded LDS stride for A (elems): 80B rows, 16B-aligned
#define LDB_S (BN + 8)     // padded LDS stride for B (elems): 272B rows, 16B-aligned

union Frag16 { v16bf v; uint4 q[2]; };

template <int EPI>
__global__ __launch_bounds__(256) void gemm_bf16_wmma(
    const unsigned short* __restrict__ A,   // [M,K] bf16
    const unsigned short* __restrict__ B,   // [K,N] bf16
    const float*          __restrict__ bias,// [N]
    float*                __restrict__ outF,
    unsigned short*       __restrict__ outB,
    int M, int N, int K)
{
    __shared__ unsigned short sA[2][BM * LDA_S];
    __shared__ unsigned short sB[2][BK * LDB_S];

    const int tid  = threadIdx.x;
    const int lane = tid & 31;
    const int wave = tid >> 5;       // 0..7
    const int wm   = wave >> 2;      // 0..1 -> 64-row slab
    const int wn   = wave & 3;       // 0..3 -> 32-col slab
    const int m0   = blockIdx.x * BM;
    const int n0   = blockIdx.y * BN;

    v8f acc[4][2];
#pragma unroll
    for (int i = 0; i < 4; ++i)
#pragma unroll
        for (int j = 0; j < 2; ++j)
#pragma unroll
            for (int r = 0; r < 8; ++r)
                acc[i][j][r] = 0.0f;

    const int khalf = (lane >> 4) * 8;                 // A frag K-half (ISA 7.12.2)
    const int krow  = (lane & 15) + (lane >> 4) * 16;  // B frag K row
    const int arow  = wm * 64 + (lane & 15);           // block-local A frag row

    // per-thread staging coords: 4 x 16B async copies per tile
    const int ar = tid >> 2, ac = (tid & 3) * 8;       // A rows 0..63 (+64)
    const int br = tid >> 4, bc = (tid & 15) * 8;      // B rows 0..15 (+16)
    const unsigned sAo[2] = { lds_addr_of(&sA[0][0]), lds_addr_of(&sA[1][0]) };
    const unsigned sBo[2] = { lds_addr_of(&sB[0][0]), lds_addr_of(&sB[1][0]) };
    const unsigned offA0 = (unsigned)(ar * LDA_S + ac) * 2;
    const unsigned offA1 = (unsigned)((ar + 64) * LDA_S + ac) * 2;
    const unsigned offB0 = (unsigned)(br * LDB_S + bc) * 2;
    const unsigned offB1 = (unsigned)((br + 16) * LDB_S + bc) * 2;

    // pointer-stepped global sources: constant stride per K-tile
    const unsigned short* aPtr0 = A + (size_t)(m0 + ar) * K + ac;
    const unsigned short* aPtr1 = A + (size_t)(m0 + ar + 64) * K + ac;
    const unsigned short* bPtr0 = B + (size_t)br * N + n0 + bc;
    const unsigned short* bPtr1 = B + (size_t)(br + 16) * N + n0 + bc;
    const size_t bStep = (size_t)BK * N;

    auto stage = [&](int b) {
        async_ld_b128(sAo[b] + offA0, aPtr0);
        async_ld_b128(sAo[b] + offA1, aPtr1);
        async_ld_b128(sBo[b] + offB0, bPtr0);
        async_ld_b128(sBo[b] + offB1, bPtr1);
        aPtr0 += BK; aPtr1 += BK; bPtr0 += bStep; bPtr1 += bStep;
    };

    const int nk = K / BK;
    stage(0);

    for (int kt = 0; kt < nk; ++kt) {
        const int cur = kt & 1;
        if (kt + 1 < nk) {
            stage(cur ^ 1);     // overlap next tile's loads with this compute
            wait_async_le4();   // first 4 (this tile) retired, in order
        } else {
            wait_async_0();
        }
        __syncthreads();

        const unsigned short* bufA = &sA[cur][0];
        const unsigned short* bufB = &sB[cur][0];

        Frag16 af[4], bf[2];
#pragma unroll
        for (int j = 0; j < 2; ++j) {
            const unsigned short* p = &bufB[khalf * LDB_S + wn * 32 + j * 16 + (lane & 15)];
            unsigned short* bq = (unsigned short*)&bf[j];
#pragma unroll
            for (int s = 0; s < 8; ++s) {
                bq[s]     = p[s * LDB_S];          // K = khalf .. khalf+7, col N=lane&15
                bq[s + 8] = p[(s + 16) * LDB_S];   // K = khalf+16 .. khalf+23
            }
        }
#pragma unroll
        for (int i = 0; i < 4; ++i) {
            const unsigned short* p = &bufA[(arow + i * 16) * LDA_S + khalf];
            af[i].q[0] = *(const uint4*)p;          // K = khalf .. khalf+7
            af[i].q[1] = *(const uint4*)(p + 16);   // K = khalf+16 .. khalf+23
        }
#pragma unroll
        for (int i = 0; i < 4; ++i)
#pragma unroll
            for (int j = 0; j < 2; ++j)
                acc[i][j] = __builtin_amdgcn_wmma_f32_16x16x32_bf16(
                    false, af[i].v, false, bf[j].v,
                    (short)0, acc[i][j], false, false);
        __syncthreads();   // protect this buffer from the next iteration's stage
    }

    // ---- epilogue (C/D layout: lane&15 = N col, (lane>>4)*8 + r = M row) ----
#pragma unroll
    for (int i = 0; i < 4; ++i) {
#pragma unroll
        for (int j = 0; j < 2; ++j) {
            int cn = n0 + wn * 32 + j * 16 + (lane & 15);
            int cm = m0 + wm * 64 + i * 16 + (lane >> 4) * 8;
            float bv = bias[cn];
#pragma unroll
            for (int r = 0; r < 8; ++r) {
                float v = acc[i][j][r] + bv;
                if (EPI == 1) {
                    outB[(size_t)(cm + r) * N + cn] = f2bf(tanhf(v));
                } else {
                    outF[(size_t)(cm + r) * N + cn] = v;
                }
            }
        }
    }
}

// ---------------------------------------------------------------------------
// LayerNorm (over last dim) + LeakyReLU(0.1). One block (256 thr) per row.
// ---------------------------------------------------------------------------
__global__ __launch_bounds__(256) void ln_lrelu_kernel(
    const float* __restrict__ in, const float* __restrict__ g,
    const float* __restrict__ beta,
    unsigned short* __restrict__ outB, float* __restrict__ outF, int N)
{
    const int row = blockIdx.x;
    const float* x = in + (size_t)row * N;
    float s = 0.f, s2 = 0.f;
    for (int i = threadIdx.x; i < N; i += 256) {
        float v = x[i];
        s += v; s2 += v * v;
    }
#pragma unroll
    for (int off = 16; off > 0; off >>= 1) {
        s  += __shfl_xor(s,  off, 32);
        s2 += __shfl_xor(s2, off, 32);
    }
    __shared__ float rs[8], rs2[8];
    int wave = threadIdx.x >> 5, lane = threadIdx.x & 31;
    if (lane == 0) { rs[wave] = s; rs2[wave] = s2; }
    __syncthreads();
    if (wave == 0) {
        s  = (lane < 8) ? rs[lane]  : 0.f;
        s2 = (lane < 8) ? rs2[lane] : 0.f;
#pragma unroll
        for (int off = 4; off > 0; off >>= 1) {
            s  += __shfl_xor(s,  off, 32);
            s2 += __shfl_xor(s2, off, 32);
        }
        if (lane == 0) { rs[0] = s; rs2[0] = s2; }
    }
    __syncthreads();
    float mean = rs[0] / N;
    float var  = rs2[0] / N - mean * mean;
    float rstd = rsqrtf(var + 1e-5f);
    for (int i = threadIdx.x; i < N; i += 256) {
        float v = (x[i] - mean) * rstd * g[i] + beta[i];
        v = (v >= 0.f) ? v : LEAKY * v;
        outB[(size_t)row * N + i] = f2bf(v);
        if (outF) outF[(size_t)row * N + i] = v;
    }
}

// ---------------------------------------------------------------------------
// Elementwise helpers
// ---------------------------------------------------------------------------
__global__ void convert_bf16_kernel(const float* __restrict__ src,
                                    unsigned short* __restrict__ dst, int n) {
    int i = blockIdx.x * 256 + threadIdx.x;
    if (i < n) dst[i] = f2bf(src[i]);
}

// src [K,N] fp32 -> dst [N,K] bf16 (tiled transpose)
__global__ __launch_bounds__(256) void transpose_bf16_kernel(
    const float* __restrict__ src, unsigned short* __restrict__ dst, int K, int N)
{
    __shared__ float tile[32][33];
    int kb = blockIdx.y * 32, nb = blockIdx.x * 32;
    int tx = threadIdx.x & 31, ty = threadIdx.x >> 5; // 32x8
    for (int r = ty; r < 32; r += 8)
        tile[r][tx] = src[(size_t)(kb + r) * N + nb + tx];
    __syncthreads();
    for (int r = ty; r < 32; r += 8)
        dst[(size_t)(nb + r) * K + kb + tx] = f2bf(tile[tx][r]);
}

// out_bf16 = bf16(z + coef*k)
__global__ void axpy_bf16_kernel(const float* __restrict__ z,
                                 const float* __restrict__ k, float coef,
                                 unsigned short* __restrict__ out, int n) {
    int i = blockIdx.x * 256 + threadIdx.x;
    if (i < n) out[i] = f2bf(z[i] + coef * k[i]);
}

// z += dt/6*(k1 + 2k2 + 2k3 + k4); also refresh bf16 snapshot
__global__ void rk4_final_kernel(float* __restrict__ z,
                                 const float* __restrict__ k1, const float* __restrict__ k2,
                                 const float* __restrict__ k3, const float* __restrict__ k4,
                                 float dt6, unsigned short* __restrict__ zb, int n) {
    int i = blockIdx.x * 256 + threadIdx.x;
    if (i < n) {
        float v = z[i] + dt6 * (k1[i] + 2.f * k2[i] + 2.f * k3[i] + k4[i]);
        z[i] = v;
        zb[i] = f2bf(v);
    }
}

// GRU gate: h = (1-zg)*n + zg*h   (gi/gh layout [B, 3S] as [r|z|n])
__global__ void gru_gate_kernel(const float* __restrict__ gi,
                                const float* __restrict__ gh,
                                float* __restrict__ h,
                                unsigned short* __restrict__ hb, int S, int n) {
    int idx = blockIdx.x * 256 + threadIdx.x;
    if (idx >= n) return;
    int b = idx / S, j = idx - b * S;
    const float* gib = gi + (size_t)b * 3 * S;
    const float* ghb = gh + (size_t)b * 3 * S;
    float r  = sigmoidf_(gib[j]         + ghb[j]);
    float zg = sigmoidf_(gib[S + j]     + ghb[S + j]);
    float nn = tanhf(gib[2 * S + j] + r * ghb[2 * S + j]);
    float hv = h[idx];
    hv = (1.f - zg) * nn + zg * hv;
    h[idx]  = hv;
    hb[idx] = f2bf(hv);
}

__global__ void zero_state_kernel(float* __restrict__ h,
                                  unsigned short* __restrict__ hb, int n) {
    int i = blockIdx.x * 256 + threadIdx.x;
    if (i < n) { h[i] = 0.f; hb[i] = 0; }
}

// split h [B, 2L] into (h[:, :L], h[:, L:]) concatenated flat
__global__ void writeout_kernel(const float* __restrict__ h,
                                float* __restrict__ out, int Bn, int L, int n) {
    int idx = blockIdx.x * 256 + threadIdx.x;  // over Bn*2L
    if (idx >= n) return;
    int S = 2 * L;
    int b = idx / S, j = idx - b * S;
    float v = h[idx];
    if (j < L) out[(size_t)b * L + j] = v;
    else       out[(size_t)Bn * L + (size_t)b * L + (j - L)] = v;
}

// ---------------------------------------------------------------------------
// Host driver
// ---------------------------------------------------------------------------
extern "C" void kernel_launch(void* const* d_in, const int* in_sizes, int n_in,
                              void* d_out, int out_size, void* d_ws, size_t ws_size,
                              hipStream_t stream) {
    (void)in_sizes; (void)n_in; (void)out_size; (void)ws_size;

    constexpr int T = 128, Bt = 512, OBS = 256, HID = 1024, STATE = 1024;
    constexpr int S3 = 3 * STATE, LAT = 512;
    const float dt = 0.25f, hdt = 0.125f, dt6 = dt / 6.0f;

    const float* xs     = (const float*)d_in[0];
    const float* obs_W  = (const float*)d_in[1];
    const float* obs_b  = (const float*)d_in[2];
    const float* obs_g  = (const float*)d_in[3];
    const float* obs_be = (const float*)d_in[4];
    const float* lat_W  = (const float*)d_in[5];
    const float* lat_b  = (const float*)d_in[6];
    const float* lat_g  = (const float*)d_in[7];
    const float* lat_be = (const float*)d_in[8];
    const float* ode_W0 = (const float*)d_in[9];
    const float* ode_b0 = (const float*)d_in[10];
    const float* ode_W1 = (const float*)d_in[11];
    const float* ode_b1 = (const float*)d_in[12];
    const float* ode_W2 = (const float*)d_in[13];
    const float* ode_b2 = (const float*)d_in[14];
    const float* gWih   = (const float*)d_in[15];
    const float* gWhh   = (const float*)d_in[16];
    const float* gbih   = (const float*)d_in[17];
    const float* gbhh   = (const float*)d_in[18];

    // workspace carve-out (256B aligned)
    char* wp = (char*)d_ws;
    auto carve = [&](size_t bytes) -> void* {
        void* r = (void*)wp;
        wp += (bytes + 255) & ~(size_t)255;
        return r;
    };
    auto* wObs  = (unsigned short*)carve((size_t)OBS * HID * 2);    // [K,N] bf16
    auto* wLat  = (unsigned short*)carve((size_t)HID * STATE * 2);
    auto* wO0   = (unsigned short*)carve((size_t)STATE * HID * 2);
    auto* wO1   = (unsigned short*)carve((size_t)HID * HID * 2);
    auto* wO2   = (unsigned short*)carve((size_t)HID * STATE * 2);
    auto* wIhT  = (unsigned short*)carve((size_t)STATE * S3 * 2);   // Wih^T [S,3S]
    auto* wHhT  = (unsigned short*)carve((size_t)STATE * S3 * 2);   // Whh^T
    auto* xt_bf = (unsigned short*)carve((size_t)Bt * OBS * 2);
    auto* tmpF  = (float*)carve((size_t)Bt * HID * 4);
    auto* hobs  = (unsigned short*)carve((size_t)Bt * HID * 2);
    auto* z_f   = (float*)carve((size_t)Bt * STATE * 4);
    auto* z_bf  = (unsigned short*)carve((size_t)Bt * STATE * 2);
    auto* zt_bf = (unsigned short*)carve((size_t)Bt * STATE * 2);
    auto* h1_bf = (unsigned short*)carve((size_t)Bt * HID * 2);
    auto* h2_bf = (unsigned short*)carve((size_t)Bt * HID * 2);
    float* kb[4];
    for (int i = 0; i < 4; ++i) kb[i] = (float*)carve((size_t)Bt * STATE * 4);
    auto* gi_f  = (float*)carve((size_t)Bt * S3 * 4);
    auto* gh_f  = (float*)carve((size_t)Bt * S3 * 4);
    auto* h_f   = (float*)carve((size_t)Bt * STATE * 4);
    auto* h_bf  = (unsigned short*)carve((size_t)Bt * STATE * 2);

    auto ew = [&](int n) { return dim3((n + 255) / 256); };

    // ---- one-time (per launch) weight conversion ----
    convert_bf16_kernel<<<ew(OBS * HID), 256, 0, stream>>>(obs_W, wObs, OBS * HID);
    convert_bf16_kernel<<<ew(HID * STATE), 256, 0, stream>>>(lat_W, wLat, HID * STATE);
    convert_bf16_kernel<<<ew(STATE * HID), 256, 0, stream>>>(ode_W0, wO0, STATE * HID);
    convert_bf16_kernel<<<ew(HID * HID), 256, 0, stream>>>(ode_W1, wO1, HID * HID);
    convert_bf16_kernel<<<ew(HID * STATE), 256, 0, stream>>>(ode_W2, wO2, HID * STATE);
    transpose_bf16_kernel<<<dim3(STATE / 32, S3 / 32), 256, 0, stream>>>(gWih, wIhT, S3, STATE);
    transpose_bf16_kernel<<<dim3(STATE / 32, S3 / 32), 256, 0, stream>>>(gWhh, wHhT, S3, STATE);

    zero_state_kernel<<<ew(Bt * STATE), 256, 0, stream>>>(h_f, h_bf, Bt * STATE);

    auto gemmN = [&](const unsigned short* A, const unsigned short* B,
                     const float* bias, float* oF, int M, int N, int K) {
        gemm_bf16_wmma<0><<<dim3(M / BM, N / BN), 256, 0, stream>>>(A, B, bias, oF, nullptr, M, N, K);
    };
    auto gemmT = [&](const unsigned short* A, const unsigned short* B,
                     const float* bias, unsigned short* oB, int M, int N, int K) {
        gemm_bf16_wmma<1><<<dim3(M / BM, N / BN), 256, 0, stream>>>(A, B, bias, nullptr, oB, M, N, K);
    };
    auto odef = [&](const unsigned short* zin, float* kout) {
        gemmT(zin,  wO0, ode_b0, h1_bf, Bt, HID,   STATE);
        gemmT(h1_bf, wO1, ode_b1, h2_bf, Bt, HID,   HID);
        gemmN(h2_bf, wO2, ode_b2, kout,  Bt, STATE, HID);
    };

    // ---- sequential scan over time ----
    for (int t = 0; t < T; ++t) {
        // obs MLP: x -> LN(LeakyReLU) -> hobs (bf16)
        convert_bf16_kernel<<<ew(Bt * OBS), 256, 0, stream>>>(
            xs + (size_t)t * Bt * OBS, xt_bf, Bt * OBS);
        gemmN(xt_bf, wObs, obs_b, tmpF, Bt, HID, OBS);
        ln_lrelu_kernel<<<Bt, 256, 0, stream>>>(tmpF, obs_g, obs_be, hobs, nullptr, HID);
        // latent MLP -> z (f32 base + bf16 snapshot)
        gemmN(hobs, wLat, lat_b, tmpF, Bt, STATE, HID);
        ln_lrelu_kernel<<<Bt, 256, 0, stream>>>(tmpF, lat_g, lat_be, z_bf, z_f, STATE);

        // RK4 ODE solve, 4 steps over [0,1]
        for (int s = 0; s < 4; ++s) {
            odef(z_bf, kb[0]);
            axpy_bf16_kernel<<<ew(Bt * STATE), 256, 0, stream>>>(z_f, kb[0], hdt, zt_bf, Bt * STATE);
            odef(zt_bf, kb[1]);
            axpy_bf16_kernel<<<ew(Bt * STATE), 256, 0, stream>>>(z_f, kb[1], hdt, zt_bf, Bt * STATE);
            odef(zt_bf, kb[2]);
            axpy_bf16_kernel<<<ew(Bt * STATE), 256, 0, stream>>>(z_f, kb[2], dt, zt_bf, Bt * STATE);
            odef(zt_bf, kb[3]);
            rk4_final_kernel<<<ew(Bt * STATE), 256, 0, stream>>>(
                z_f, kb[0], kb[1], kb[2], kb[3], dt6, z_bf, Bt * STATE);
        }

        // GRU update
        gemmN(z_bf, wIhT, gbih, gi_f, Bt, S3, STATE);
        gemmN(h_bf, wHhT, gbhh, gh_f, Bt, S3, STATE);
        gru_gate_kernel<<<ew(Bt * STATE), 256, 0, stream>>>(gi_f, gh_f, h_f, h_bf, STATE, Bt * STATE);
    }

    writeout_kernel<<<ew(Bt * STATE), 256, 0, stream>>>(h_f, (float*)d_out, Bt, LAT, Bt * STATE);
}
